// GroupedQueryAttention_62715112456518
// MI455X (gfx1250) — compile-verified, matrix operand order checked by reading, not run on hardware
//
#include <hip/hip_runtime.h>
#include <hip/hip_bf16.h>

// ---------------------------------------------------------------------------
// GQA for MI455X (gfx1250): all matmuls through v_wmma_f32_16x16x32_f16.
// B=2, L=2048, D_MODEL=2048, H=16, KV=4, GROUP=4, D_K=128, D_V=512
// ---------------------------------------------------------------------------

#define L_SEQ   2048
#define DMODEL  2048
#define NHEADS  16
#define NKV     4
#define DK      128
#define DV      512
#define QDIM    2048   // NHEADS*DK
#define KDIM    512    // NKV*DK
#define VDIM    2048   // NKV*DV
#define QKVN    4608   // QDIM+KDIM+VDIM
#define CTXDIM  8192   // NHEADS*DV

typedef __attribute__((ext_vector_type(16))) _Float16 v16h;
typedef __attribute__((ext_vector_type(8)))  _Float16 v8h;
typedef __attribute__((ext_vector_type(8)))  float    v8f;
typedef __attribute__((ext_vector_type(4)))  _Float16 v4h;

union Frag16 { v16h h; v8h v2[2]; };

// 16x32 f16 A/B^T fragment (CDNA5 ISA 7.12.2): lanes 0-15 row=lane,
// K {0-7 -> VGPR0-3, 16-23 -> VGPR4-7}; lanes 16-31 row=lane-16, K {8-15, 24-31}.
// Exactly two 16-byte loads per fragment.
__device__ __forceinline__ v16h load_frag(const _Float16* base, int stride, int lane) {
  const int r  = lane & 15;
  const int hf = (lane >> 4) & 1;
  const _Float16* row = base + r * stride + hf * 8;
  Frag16 f;
  f.v2[0] = *(const v8h*)(row);        // K = hf*8 .. hf*8+7
  f.v2[1] = *(const v8h*)(row + 16);   // K = 16 + hf*8 .. +7
  return f.h;
}

__device__ __forceinline__ v8f wmma16(v16h a, v16h b, v8f c) {
  return __builtin_amdgcn_wmma_f32_16x16x32_f16(false, a, false, b, (short)0, c,
                                                false, false);
}

// ---------------------------------------------------------------------------
// 1a) f32 -> f16 convert (vectorized x4)
// ---------------------------------------------------------------------------
__global__ void cvt_f32_to_f16(const float* __restrict__ in,
                               _Float16* __restrict__ out, int n4) {
  int i = blockIdx.x * blockDim.x + threadIdx.x;
  if (i < n4) {
    float4 v = *(const float4*)(in + (size_t)i * 4);
    v4h h;
    h[0] = (_Float16)v.x; h[1] = (_Float16)v.y;
    h[2] = (_Float16)v.z; h[3] = (_Float16)v.w;
    *(v4h*)(out + (size_t)i * 4) = h;
  }
}

// ---------------------------------------------------------------------------
// 1b) f32 -> f16 convert WITH transpose: in[K][N] f32 -> out[N][K] f16.
//     One-time cost; removes all B-tile scatter from the hot GEMM loop.
// ---------------------------------------------------------------------------
__global__ void __launch_bounds__(256)
cvt_transpose_f32_to_f16(const float* __restrict__ in, _Float16* __restrict__ out,
                         int K, int N) {
  __shared__ _Float16 t[32][40];
  const int c0 = blockIdx.x * 32;        // input col  (= output row block)
  const int r0 = blockIdx.y * 32;        // input row  (= output col block)
  const int tx = threadIdx.x, ty = threadIdx.y;
#pragma unroll
  for (int i = 0; i < 4; ++i) {
    int r = ty + i * 8;
    t[r][tx] = (_Float16)in[(size_t)(r0 + r) * N + c0 + tx];   // coalesced read
  }
  __syncthreads();
#pragma unroll
  for (int i = 0; i < 4; ++i) {
    int rr = ty + i * 8;
    out[(size_t)(c0 + rr) * K + r0 + tx] = t[tx][rr];          // coalesced write
  }
}

// ---------------------------------------------------------------------------
// 2) WMMA GEMM: C(f32)[M][N] = A(f16)[M][K] * BT(f16)[N][K] (+bias)
//    256 threads = 8 waves; 128x128 block tile; K step 64; wave tile 32x64.
//    Both A and BT tiles stage with pure b128 copies (no scatter).
// ---------------------------------------------------------------------------
__global__ void __launch_bounds__(256)
gemm_wmma_f16(const _Float16* __restrict__ A, const _Float16* __restrict__ BT,
              float* __restrict__ C, const float* __restrict__ bias,
              int M, int N, int K) {
  __shared__ _Float16 As[128][72];     // 128 M-rows x 64 K (padded)  18432 B
  __shared__ _Float16 Bts[128][72];    // 128 N-rows x 64 K (padded)  18432 B

  const int tid  = threadIdx.x;
  const int wave = tid >> 5, lane = tid & 31;
  const int wm = wave >> 1, wn = wave & 1;       // 4 x 2 wave grid
  const int n0 = blockIdx.x * 128, m0 = blockIdx.y * 128;

  v8f acc[2][4] = {};

  for (int k0 = 0; k0 < K; k0 += 64) {
    // ---- load phase: everything into registers first (clauses, no LDS waits)
    v8h ta[4], tb[4];
#pragma unroll
    for (int i = 0; i < 4; ++i) {
      int idx = i * 256 + tid;                 // 0..1023
      int r = idx >> 3, c8 = idx & 7;          // 128 rows x 8 v8h
      ta[i] = *(const v8h*)&A[(size_t)(m0 + r) * K + k0 + c8 * 8];
    }
#pragma unroll
    for (int i = 0; i < 4; ++i) {
      int idx = i * 256 + tid;
      int n = idx >> 3, c8 = idx & 7;          // 128 N-rows x 8 v8h
      tb[i] = *(const v8h*)&BT[(size_t)(n0 + n) * K + k0 + c8 * 8];
    }
    if (k0 + 64 < K) {                         // gfx1250 global_prefetch
      __builtin_prefetch(A + (size_t)(m0 + (tid & 127)) * K + k0 + 64, 0, 1);
      __builtin_prefetch(BT + (size_t)(n0 + (tid & 127)) * K + k0 + 64, 0, 1);
    }
    __syncthreads();                            // prev iter consumers done
    // ---- store phase: straight b128 stores for both tiles
#pragma unroll
    for (int i = 0; i < 4; ++i) {
      int idx = i * 256 + tid;
      int r = idx >> 3, c8 = idx & 7;
      *(v8h*)&As[r][c8 * 8] = ta[i];
    }
#pragma unroll
    for (int i = 0; i < 4; ++i) {
      int idx = i * 256 + tid;
      int n = idx >> 3, c8 = idx & 7;
      *(v8h*)&Bts[n][c8 * 8] = tb[i];
    }
    __syncthreads();

    // ---- compute: 16 WMMAs per wave per K-chunk
#pragma unroll
    for (int kc = 0; kc < 2; ++kc) {
      v16h af0 = load_frag(&As[wm * 32 +  0][kc * 32], 72, lane);
      v16h af1 = load_frag(&As[wm * 32 + 16][kc * 32], 72, lane);
#pragma unroll
      for (int nt = 0; nt < 4; ++nt) {
        v16h bf = load_frag(&Bts[wn * 64 + nt * 16][kc * 32], 72, lane);
        acc[0][nt] = wmma16(af0, bf, acc[0][nt]);
        acc[1][nt] = wmma16(af1, bf, acc[1][nt]);
      }
    }
  }

  // epilogue: C layout -> lanes 0-15 rows 0-7, lanes 16-31 rows 8-15
  const int nn = lane & 15, roff = (lane >> 4) << 3;
#pragma unroll
  for (int mt = 0; mt < 2; ++mt)
#pragma unroll
    for (int nt = 0; nt < 4; ++nt) {
      int col = n0 + wn * 64 + nt * 16 + nn;
      float bv = bias ? bias[col] : 0.0f;
#pragma unroll
      for (int r = 0; r < 8; ++r) {
        int row = m0 + wm * 32 + mt * 16 + roff + r;
        C[(size_t)row * N + col] = acc[mt][nt][r] + bv;
      }
    }
}

// ---------------------------------------------------------------------------
// 3) RoPE + split/convert: qkv(f32) -> qh(B,H,L,DK) f16 (pre-scaled 1/sqrt(DK)),
//    kh(B,KV,L,DK) f16, vT(B,KV,DV,L) f16  (V TRANSPOSED for the PV WMMA)
// ---------------------------------------------------------------------------
__global__ void rope_split(const float* __restrict__ qkv,
                           _Float16* __restrict__ qh,
                           _Float16* __restrict__ kh,
                           _Float16* __restrict__ vT) {
  const int row = blockIdx.x;               // b*L + l
  const int b = row >> 11, l = row & (L_SEQ - 1);
  const float* base = qkv + (size_t)row * QKVN;
  const int tid = threadIdx.x;
  const float kLogTheta = 9.210340371976184f;   // ln(10000)
  const float kQScale   = 0.08838834764831845f; // 1/sqrt(128)

  for (int idx = tid; idx < QDIM; idx += 256) {
    int h = idx >> 7, j = idx & 127, jm = j & 63;
    float inv = __expf(-kLogTheta * (float)(2 * jm) * (1.0f / 128.0f));
    float s, c;
    __sincosf((float)l * inv, &s, &c);
    const float* qr = base + h * DK;
    float xj  = qr[j];
    float rot = (j < 64) ? -qr[j + 64] : qr[j - 64];
    qh[((size_t)(b * NHEADS + h) * L_SEQ + l) * DK + j] =
        (_Float16)((xj * c + rot * s) * kQScale);
  }
  for (int idx = tid; idx < KDIM; idx += 256) {
    int kvh = idx >> 7, j = idx & 127, jm = j & 63;
    float inv = __expf(-kLogTheta * (float)(2 * jm) * (1.0f / 128.0f));
    float s, c;
    __sincosf((float)l * inv, &s, &c);
    const float* kr = base + QDIM + kvh * DK;
    float xj  = kr[j];
    float rot = (j < 64) ? -kr[j + 64] : kr[j - 64];
    kh[((size_t)(b * NKV + kvh) * L_SEQ + l) * DK + j] =
        (_Float16)(xj * c + rot * s);
  }
  // v: convert + transpose -> vT[b][kv][e][l]
  for (int idx = tid; idx < VDIM; idx += 256) {
    int kvh = idx >> 9, e = idx & 511;
    vT[((size_t)(b * NKV + kvh) * DV + e) * L_SEQ + l] =
        (_Float16)base[QDIM + KDIM + idx];
  }
}

// ---------------------------------------------------------------------------
// 4) Flash attention: grid (L/128, DV/128, B*NHEADS), block 256 (8 waves).
//    Each wave: 16 q-rows x 128 v-cols; streams 64-key tiles.
// ---------------------------------------------------------------------------
__global__ void __launch_bounds__(256)
flash_attn(const _Float16* __restrict__ qh, const _Float16* __restrict__ kh,
           const _Float16* __restrict__ vT, _Float16* __restrict__ ctx) {
  __shared__ _Float16 Ks[64][136];     // keys x DK          17408 B
  __shared__ _Float16 Vt[128][72];     // v-cols x keys      18432 B
  __shared__ _Float16 Pb[8][16][72];   // per-wave P tiles   18432 B

  const int tid  = threadIdx.x;
  const int wave = tid >> 5, lane = tid & 31;
  const int qrow0 = blockIdx.x * 128;
  const int vcol0 = blockIdx.y * 128;
  const int bz = blockIdx.z;
  const int b = bz >> 4, h = bz & 15;
  const int kv = h >> 2;               // GROUP = 4

  // q fragments in registers: 16 rows x DK=128 (4 K-chunks of 32)
  const _Float16* qbase =
      qh + ((size_t)(b * NHEADS + h) * L_SEQ + qrow0 + wave * 16) * DK;
  v16h qf[4];
#pragma unroll
  for (int kc = 0; kc < 4; ++kc) qf[kc] = load_frag(qbase + kc * 32, DK, lane);

  float m_i[8], l_i[8];
  v8f oacc[8] = {};
#pragma unroll
  for (int r = 0; r < 8; ++r) { m_i[r] = -1e30f; l_i[r] = 0.0f; }

  const _Float16* kg = kh + (size_t)(b * NKV + kv) * L_SEQ * DK;
  const _Float16* vg = vT + ((size_t)(b * NKV + kv) * DV + vcol0) * L_SEQ;

  for (int kt0 = 0; kt0 < L_SEQ; kt0 += 64) {
    // ---- stage K tile 64 x 128 (row-major == B^T for S = q @ k^T)
    v8h tk[4];
#pragma unroll
    for (int i = 0; i < 4; ++i) {
      int idx = i * 256 + tid;                  // 0..1023
      int r = idx >> 4, c8 = idx & 15;
      tk[i] = *(const v8h*)&kg[(size_t)(kt0 + r) * DK + c8 * 8];
    }
    __syncthreads();                            // prev tile consumers done
#pragma unroll
    for (int i = 0; i < 4; ++i) {
      int idx = i * 256 + tid;
      int r = idx >> 4, c8 = idx & 15;
      *(v8h*)&Ks[r][c8 * 8] = tk[i];
    }
    // ---- stage V tile from pre-transposed vT: straight b128 copies
    v8h tv[4];
#pragma unroll
    for (int i = 0; i < 4; ++i) {
      int idx = i * 256 + tid;                  // 0..1023
      int c = idx >> 3, k8 = idx & 7;
      tv[i] = *(const v8h*)&vg[(size_t)c * L_SEQ + kt0 + k8 * 8];
    }
#pragma unroll
    for (int i = 0; i < 4; ++i) {
      int idx = i * 256 + tid;
      int c = idx >> 3, k8 = idx & 7;
      *(v8h*)&Vt[c][k8 * 8] = tv[i];
    }
    __syncthreads();

    // ---- S = q @ k^T : 16 rows x 64 keys per wave (16 WMMAs)
    v8f s[4] = {};
#pragma unroll
    for (int kc = 0; kc < 4; ++kc)
#pragma unroll
      for (int nt = 0; nt < 4; ++nt) {
        v16h bf = load_frag(&Ks[nt * 16][kc * 32], 136, lane);
        s[nt] = wmma16(qf[kc], bf, s[nt]);
      }

    // ---- online softmax: row stats replicated within each 16-lane half
    float cm[8], rs[8], sc[8];
#pragma unroll
    for (int r = 0; r < 8; ++r)
      cm[r] = fmaxf(fmaxf(s[0][r], s[1][r]), fmaxf(s[2][r], s[3][r]));
#pragma unroll
    for (int mask = 1; mask <= 8; mask <<= 1)
#pragma unroll
      for (int r = 0; r < 8; ++r)
        cm[r] = fmaxf(cm[r], __shfl_xor(cm[r], mask, 32));
#pragma unroll
    for (int r = 0; r < 8; ++r) {
      float nm = fmaxf(m_i[r], cm[r]);
      sc[r] = __expf(m_i[r] - nm);
      m_i[r] = nm;
      rs[r] = 0.0f;
    }
#pragma unroll
    for (int nt = 0; nt < 4; ++nt)
#pragma unroll
      for (int r = 0; r < 8; ++r) {
        float p = __expf(s[nt][r] - m_i[r]);
        s[nt][r] = p;
        rs[r] += p;
      }
#pragma unroll
    for (int mask = 1; mask <= 8; mask <<= 1)
#pragma unroll
      for (int r = 0; r < 8; ++r) rs[r] += __shfl_xor(rs[r], mask, 32);
#pragma unroll
    for (int r = 0; r < 8; ++r) l_i[r] = l_i[r] * sc[r] + rs[r];
#pragma unroll
    for (int nt = 0; nt < 8; ++nt)
#pragma unroll
      for (int r = 0; r < 8; ++r) oacc[nt][r] *= sc[r];

    // ---- C-layout -> A-layout for P via per-wave LDS tile
    {
      const int nn = lane & 15, roff = (lane >> 4) << 3;
#pragma unroll
      for (int nt = 0; nt < 4; ++nt)
#pragma unroll
        for (int r = 0; r < 8; ++r)
          Pb[wave][roff + r][nt * 16 + nn] = (_Float16)s[nt][r];
    }
    __syncthreads();

    // ---- O += P @ V : 2 K-chunks x 8 col-tiles (16 WMMAs)
#pragma unroll
    for (int kc2 = 0; kc2 < 2; ++kc2) {
      v16h pf = load_frag(&Pb[wave][0][kc2 * 32], 72, lane);
#pragma unroll
      for (int nt = 0; nt < 8; ++nt) {
        v16h bf = load_frag(&Vt[nt * 16][kc2 * 32], 72, lane);
        oacc[nt] = wmma16(pf, bf, oacc[nt]);
      }
    }
  }

  // epilogue: normalize and write ctx (B, L, H*DV) f16
  const int nn = lane & 15, roff = (lane >> 4) << 3;
  _Float16* cbase = ctx + ((size_t)b * L_SEQ + qrow0 + wave * 16) * CTXDIM +
                    (size_t)h * DV + vcol0;
#pragma unroll
  for (int nt = 0; nt < 8; ++nt)
#pragma unroll
    for (int r = 0; r < 8; ++r)
      cbase[(size_t)(roff + r) * CTXDIM + nt * 16 + nn] =
          (_Float16)(oacc[nt][r] / l_i[r]);
}

// ---------------------------------------------------------------------------
// Launch: x, W_attn, W_out, b_out  ->  out (f32)
// ---------------------------------------------------------------------------
extern "C" void kernel_launch(void* const* d_in, const int* in_sizes, int n_in,
                              void* d_out, int out_size, void* d_ws, size_t ws_size,
                              hipStream_t stream) {
  (void)in_sizes; (void)n_in; (void)out_size; (void)ws_size;
  const float* x      = (const float*)d_in[0];
  const float* W_attn = (const float*)d_in[1];
  const float* W_out  = (const float*)d_in[2];
  const float* b_out  = (const float*)d_in[3];
  float* out = (float*)d_out;

  const int M = 2 * L_SEQ;  // 4096 rows (B*L)

  // workspace layout (bytes)
  char* ws = (char*)d_ws;
  _Float16* Xh   = (_Float16*)(ws);                       // 16,777,216
  _Float16* WahT = (_Float16*)(ws + 16777216);            // 18,874,368 (QKVN x DMODEL)
  _Float16* WohT = (_Float16*)(ws + 35651584);            // 33,554,432 (DMODEL x CTXDIM)
  float*    qkv  = (float*)   (ws + 69206016);            // 75,497,472
  _Float16* ctxh = (_Float16*)(ws + 69206016);            // aliases qkv (dead after rope)
  _Float16* qhp  = (_Float16*)(ws + 144703488);           // 16,777,216
  _Float16* khp  = (_Float16*)(ws + 161480704);           //  4,194,304
  _Float16* vTp  = (_Float16*)(ws + 165675008);           // 16,777,216

  // 1) convert inputs to f16 (weights transposed once -> scatter-free GEMMs)
  {
    int n4 = (M * DMODEL) / 4;
    cvt_f32_to_f16<<<(n4 + 255) / 256, 256, 0, stream>>>(x, Xh, n4);
    cvt_transpose_f32_to_f16<<<dim3(QKVN / 32, DMODEL / 32), dim3(32, 8), 0,
                               stream>>>(W_attn, WahT, DMODEL, QKVN);
    cvt_transpose_f32_to_f16<<<dim3(DMODEL / 32, CTXDIM / 32), dim3(32, 8), 0,
                               stream>>>(W_out, WohT, CTXDIM, DMODEL);
  }

  // 2) qkv = x @ W_attn   (4096 x 4608, K=2048)
  gemm_wmma_f16<<<dim3(QKVN / 128, M / 128), 256, 0, stream>>>(
      Xh, WahT, qkv, nullptr, M, QKVN, DMODEL);

  // 3) RoPE + head split (+ V transpose)
  rope_split<<<M, 256, 0, stream>>>(qkv, qhp, khp, vTp);

  // 4) attention -> ctx (f16)
  flash_attn<<<dim3(L_SEQ / 128, DV / 128, 2 * NHEADS), 256, 0, stream>>>(
      qhp, khp, vTp, ctxh);

  // 5) out = ctx @ W_out + b_out   (4096 x 2048, K=8192)
  gemm_wmma_f16<<<dim3(DMODEL / 128, M / 128), 256, 0, stream>>>(
      ctxh, WohT, out, b_out, M, DMODEL, CTXDIM);
}
